// TimeAttentionLayer_31696858644744
// MI455X (gfx1250) — matrix operand order repaired and checked by reading, not compiled, not run on hardware
//
#include <hip/hip_runtime.h>
#include <hip/hip_bf16.h>

// ---------------- problem constants ----------------
#define BB 4
#define NN 128
#define LL 64
#define DD 256
#define HH 8
#define DKK 32
#define MROWS (BB * NN * LL)           // 32768
#define NEGV (-1000000000.0f)
#define LN_EPS 1e-5f

// GEMM dynamic LDS: A panel 64x256 f32 (64KB) + double-buffered W tile 2x(256x32) f16 (32KB)
#define GEMM_LDS_A_BYTES  (64 * 256 * 4)
#define GEMM_LDS_BYTES    (GEMM_LDS_A_BYTES + 2 * 256 * 32 * 2)

typedef _Float16 half_t;
typedef __bf16   bf16_t;
typedef __attribute__((ext_vector_type(16))) _Float16 v16h;
typedef __attribute__((ext_vector_type(16))) __bf16   v16bf;
typedef __attribute__((ext_vector_type(8)))  float    v8f;
typedef __attribute__((ext_vector_type(4)))  unsigned int v4u;
typedef __attribute__((ext_vector_type(8)))  int      v8i;
typedef __attribute__((ext_vector_type(4)))  int      v4i;

#define DEV static __device__ __forceinline__

// ---------------- WMMA wrappers ----------------
DEV v8f wmma_f16(v16h a, v16h b, v8f c) {
  return __builtin_amdgcn_wmma_f32_16x16x32_f16(false, a, false, b, (short)0, c, false, false);
}
DEV v8f wmma_bf16(v16bf a, v16bf b, v8f c) {
  return __builtin_amdgcn_wmma_f32_16x16x32_bf16(false, a, false, b, (short)0, c, false, false);
}

// ---------------- TDM: 2D tile load Global -> LDS ----------------
// Packs a D# per CDNA5 ISA (group0 + group1, 2D tile,
// groups 2/3 zero). data_size code: 0=1B,1=2B,2=4B.
DEV void tdm_load_2d(unsigned lds_addr, const void* gptr, unsigned ds_code,
                     unsigned tensor_d0, unsigned tensor_d1,
                     unsigned tile_d0, unsigned tile_d1, unsigned stride0) {
  unsigned long long ga = (unsigned long long)gptr;
  v4u g0;
  g0[0] = 1u;                                               // count=1 (valid user D#)
  g0[1] = lds_addr;                                         // LDS byte address
  g0[2] = (unsigned)(ga & 0xffffffffu);                     // global_addr[31:0]
  g0[3] = (unsigned)((ga >> 32) & 0x01ffffffu) | (2u << 30);// global_addr[56:32] | type=2
  v8i g1;
  g1[0] = (int)(ds_code << 16);                             // wg_mask=0, data_size
  g1[1] = (int)((tensor_d0 & 0xffffu) << 16);               // dim0[15:0] @ bits[63:48]
  g1[2] = (int)(((tensor_d0 >> 16) & 0xffffu) | ((tensor_d1 & 0xffffu) << 16));
  g1[3] = (int)(((tensor_d1 >> 16) & 0xffffu) | ((tile_d0 & 0xffffu) << 16));
  g1[4] = (int)(tile_d1 & 0xffffu);                         // tile_dim1, tile_dim2=0
  g1[5] = (int)stride0;                                     // dim0_stride[31:0]
  g1[6] = 0;                                                // dim0_stride[47:32], dim1_stride lo
  g1[7] = 0;
  v4i z4 = {0, 0, 0, 0};
#if __clang_major__ >= 23
  v8i z8 = {0, 0, 0, 0, 0, 0, 0, 0};
  __builtin_amdgcn_tensor_load_to_lds(g0, g1, z4, z4, z8, 0);
#else
  __builtin_amdgcn_tensor_load_to_lds(g0, g1, z4, z4, 0);
#endif
}

// ---------------- async Global -> LDS (16B per lane) ----------------
DEV void async_ld_b128(void* lds, const void* g) {
  unsigned l = (unsigned)(unsigned long long)lds;           // low 32 bits = LDS offset
  unsigned long long ga = (unsigned long long)g;
  asm volatile("global_load_async_to_lds_b128 %0, %1, off" :: "v"(l), "v"(ga) : "memory");
}
DEV void async_wait0() { asm volatile("s_wait_asynccnt 0x0" ::: "memory"); }

// ---------------- fragment loaders (from LDS, row-major [outer][K]) ----------------
// A-matrix 16x32: lanes 0-15: M=lane, K pairs {0..7,16..23}; lanes 16-31: {8..15,24..31}
DEV v16h frag_a_h(const half_t* base, int stride, int lane) {
  const half_t* row = base + (lane & 15) * stride;
  int kb = (lane < 16) ? 0 : 8;
  v16h f;
#pragma unroll
  for (int i = 0; i < 8; ++i) {
    int kk = ((i & 4) ? 16 : 0) + kb + 2 * (i & 3);
    f[2 * i] = row[kk];
    f[2 * i + 1] = row[kk + 1];
  }
  return f;
}
// same A layout but source tile kept in f32 (TDM can't convert) -> cvt at build
DEV v16h frag_a_f32h(const float* base, int stride, int lane) {
  const float* row = base + (lane & 15) * stride;
  int kb = (lane < 16) ? 0 : 8;
  v16h f;
#pragma unroll
  for (int i = 0; i < 8; ++i) {
    int kk = ((i & 4) ? 16 : 0) + kb + 2 * (i & 3);
    f[2 * i] = (half_t)row[kk];
    f[2 * i + 1] = (half_t)row[kk + 1];
  }
  return f;
}
DEV v16bf frag_a_bf(const bf16_t* base, int stride, int lane) {
  const bf16_t* row = base + (lane & 15) * stride;
  int kb = (lane < 16) ? 0 : 8;
  v16bf f;
#pragma unroll
  for (int i = 0; i < 8; ++i) {
    int kk = ((i & 4) ? 16 : 0) + kb + 2 * (i & 3);
    f[2 * i] = row[kk];
    f[2 * i + 1] = row[kk + 1];
  }
  return f;
}
// B-matrix 32x16 via N-major [N][K] tile: lane=N; lanes 0-15: K {0..7,16..23}, 16-31: K {8..15,24..31}
DEV v16h frag_b_h(const half_t* base, int stride, int lane) {
  const half_t* row = base + (lane & 15) * stride;
  int kb = (lane < 16) ? 0 : 8;
  v16h f;
#pragma unroll
  for (int i = 0; i < 8; ++i) {
    int kk = ((i & 4) ? 16 : 0) + kb + 2 * (i & 3);
    f[2 * i] = row[kk];
    f[2 * i + 1] = row[kk + 1];
  }
  return f;
}
DEV v16bf frag_b_bf(const bf16_t* base, int stride, int lane) {
  const bf16_t* row = base + (lane & 15) * stride;
  int kb = (lane < 16) ? 0 : 8;
  v16bf f;
#pragma unroll
  for (int i = 0; i < 8; ++i) {
    int kk = ((i & 4) ? 16 : 0) + kb + 2 * (i & 3);
    f[2 * i] = row[kk];
    f[2 * i + 1] = row[kk + 1];
  }
  return f;
}

// ======================================================================
// Weight prep: Wt[n][k] = (f16) W[k][n]   (256 x 256)
// ======================================================================
__global__ __launch_bounds__(256)
void ta_prep_wt(const float* __restrict__ W, half_t* __restrict__ Wt) {
  int n = blockIdx.x;
  int k = threadIdx.x;
  Wt[n * 256 + k] = (half_t)W[k * 256 + n];
}

// ======================================================================
// GEMM: C[M x 256] = A[M x 256] @ W[256 x 256] + bias
// Block = 64 rows x all 256 cols, 256 threads (8 waves), wave w owns cols
// [w*32, w*32+32). A panel (64x256 f32) loaded ONCE by a single TDM
// descriptor; 256x32 f16 weight tiles double-buffered via TDM so the DMA
// of step k+1 overlaps the WMMAs of step k.
// ======================================================================
__global__ __launch_bounds__(256)
void ta_gemm256(const float* __restrict__ A, const half_t* __restrict__ Wt,
                const float* __restrict__ bias, float* __restrict__ C) {
  extern __shared__ __align__(16) char dynLds[];
  float*  Asf = (float*)dynLds;                         // [64][256] f32
  half_t* Wsh = (half_t*)(dynLds + GEMM_LDS_A_BYTES);   // [2][256][32] f16 (n-major)

  const int t = threadIdx.x;
  const int lane = t & 31;
  const int wv = t >> 5;
  const int m0 = blockIdx.x * 64;
  const int nw = wv * 32;          // wave's output column base

  v8f acc[4][2];
#pragma unroll
  for (int i = 0; i < 4; ++i)
#pragma unroll
    for (int j = 0; j < 2; ++j) acc[i][j] = (v8f){0, 0, 0, 0, 0, 0, 0, 0};

  if (wv == 0) {
    // whole A panel: rows m0..m0+63, 256 cols, one descriptor
    tdm_load_2d((unsigned)(unsigned long long)(void*)Asf,
                A + (size_t)m0 * 256, /*4B*/ 2u,
                256u, (unsigned)MROWS, 256u, 64u, 256u);
    // first W tile: all 256 n-rows, k=0..31 of transposed f16 weights
    tdm_load_2d((unsigned)(unsigned long long)(void*)Wsh,
                Wt, /*2B*/ 1u, 256u, 256u, 32u, 256u, 256u);
    __builtin_amdgcn_s_wait_tensorcnt((short)0);
  }
  __syncthreads();

  for (int ks = 0; ks < 8; ++ks) {
    const int k0 = ks * 32;
    if (wv == 0 && ks < 7) {
      // prefetch next W tile into the other buffer (overlaps WMMA below)
      tdm_load_2d((unsigned)(unsigned long long)(void*)(Wsh + ((ks + 1) & 1) * (256 * 32)),
                  Wt + (k0 + 32), /*2B*/ 1u, 256u, 256u, 32u, 256u, 256u);
    }
    const half_t* Ws = Wsh + (ks & 1) * (256 * 32);
    v16h b0 = frag_b_h(Ws + nw * 32, 32, lane);
    v16h b1 = frag_b_h(Ws + (nw + 16) * 32, 32, lane);
#pragma unroll
    for (int mt = 0; mt < 4; ++mt) {
      v16h a = frag_a_f32h(Asf + (mt * 16) * 256 + k0, 256, lane);
      acc[mt][0] = wmma_f16(a, b0, acc[mt][0]);
      acc[mt][1] = wmma_f16(a, b1, acc[mt][1]);
    }
    if (wv == 0) __builtin_amdgcn_s_wait_tensorcnt((short)0);
    __syncthreads();
  }

  const int rhalf = (lane < 16) ? 0 : 8;
  const int col0 = nw + (lane & 15);
  const int col1 = col0 + 16;
  const float bi0 = bias[col0];
  const float bi1 = bias[col1];
#pragma unroll
  for (int mt = 0; mt < 4; ++mt) {
#pragma unroll
    for (int i = 0; i < 8; ++i) {
      size_t gr = (size_t)(m0 + mt * 16 + rhalf + i) * 256;
      C[gr + col0] = acc[mt][0][i] + bi0;
      C[gr + col1] = acc[mt][1][i] + bi1;
    }
  }
}

// ======================================================================
// Attention: one block per (b,n,h); 128 threads = 4 waves, 16 query rows/wave.
// ======================================================================
__global__ __launch_bounds__(128)
void ta_attn(const float* __restrict__ TQ, const float* __restrict__ TK,
             const float* __restrict__ TV, const float* __restrict__ GQ,
             const float* __restrict__ GK, const float* __restrict__ GV,
             const unsigned char* __restrict__ mask,
             float* __restrict__ ts_out, float* __restrict__ gs_out,
             float* __restrict__ ms_out,
             float* __restrict__ TVAL, float* __restrict__ MVAL) {
  __shared__ half_t sTQ[64 * 32], sTK[64 * 32], sGQ[64 * 32], sGK[64 * 32];
  __shared__ bf16_t sTVt[32 * 64], sGVt[32 * 64];   // [d][key] (transposed V)
  __shared__ float  sMerge[64 * 64];
  __shared__ bf16_t sTS[64 * 64];                   // masked time_score (bf16 holds -1e9)
  __shared__ bf16_t sMS[64 * 64];                   // softmaxed merge score
  __shared__ unsigned char sMask[64 * 64];          // async-copied mask tile

  const int blk = blockIdx.x;       // (b*N+n)*H + h
  const int h = blk & (HH - 1);
  const int bn = blk >> 3;
  const int t = threadIdx.x, lane = t & 31, wv = t >> 5;

  const size_t featBase = (size_t)bn * LL * DD + (size_t)h * DKK;  // + l*DD + d
  const size_t maskBase = (size_t)bn * LL * LL;

  // ---- async-copy mask tile (4KB, no conversion needed) ----
  {
    const unsigned char* gm = mask + maskBase;
#pragma unroll
    for (int j = 0; j < 2; ++j) {
      int c = (t * 2 + j) * 16;
      async_ld_b128(&sMask[c], gm + c);
    }
    async_wait0();
  }
  // ---- stage Q/K (f16) and transposed V (bf16) ----
  {
    int e0 = t * 16;
    int r = e0 >> 5, c = e0 & 31;           // 16 contiguous elements within a row
    size_t g = featBase + (size_t)r * DD + c;
#pragma unroll
    for (int j = 0; j < 16; ++j) {
      sTQ[r * 32 + c + j] = (half_t)TQ[g + j];
      sTK[r * 32 + c + j] = (half_t)TK[g + j];
      sGQ[r * 32 + c + j] = (half_t)GQ[g + j];
      sGK[r * 32 + c + j] = (half_t)GK[g + j];
      sTVt[(c + j) * 64 + r] = (bf16_t)TV[g + j];
      sGVt[(c + j) * 64 + r] = (bf16_t)GV[g + j];
    }
  }
  __syncthreads();

  const float scale = 0.17677669529663687f;  // 1/sqrt(32)
  const int m = wv * 16;
  const size_t scoreBase = (size_t)blk * LL * LL;

  // ---- scores: Q @ K^T (K=32, one WMMA per 16x16 tile) ----
  v16h aT = frag_a_h(sTQ + m * 32, 32, lane);
  v16h aG = frag_a_h(sGQ + m * 32, 32, lane);
#pragma unroll
  for (int ntile = 0; ntile < 4; ++ntile) {
    v16h bT = frag_b_h(sTK + ntile * 16 * 32, 32, lane);  // lane=key, regs=d
    v16h bG = frag_b_h(sGK + ntile * 16 * 32, 32, lane);
    v8f z0 = {0, 0, 0, 0, 0, 0, 0, 0};
    v8f z1 = {0, 0, 0, 0, 0, 0, 0, 0};
    v8f accT = wmma_f16(aT, bT, z0);
    v8f accG = wmma_f16(aG, bG, z1);
    const int q0 = m + ((lane < 16) ? 0 : 8);
    const int kc = ntile * 16 + (lane & 15);
#pragma unroll
    for (int i = 0; i < 8; ++i) {
      int q = q0 + i;
      bool mk = sMask[q * 64 + kc] != 0;
      float ts = mk ? NEGV : accT[i] * scale;
      float gs = mk ? NEGV : accG[i] * scale;
      float ms = mk ? NEGV : (ts + gs);
      ts_out[scoreBase + (size_t)q * LL + kc] = ts;
      gs_out[scoreBase + (size_t)q * LL + kc] = gs;
      sTS[q * 64 + kc] = (bf16_t)ts;
      sMerge[q * 64 + kc] = ms;
    }
  }
  __syncthreads();

  // ---- softmax over merge rows (one thread per query row) ----
  if (t < 64) {
    float* row = sMerge + t * 64;
    float mx = row[0];
#pragma unroll 4
    for (int k = 1; k < 64; ++k) mx = fmaxf(mx, row[k]);
    float s = 0.f;
#pragma unroll 4
    for (int k = 0; k < 64; ++k) { float e = __expf(row[k] - mx); row[k] = e; s += e; }
    float inv = 1.f / s;
#pragma unroll 4
    for (int k = 0; k < 64; ++k) {
      float p = row[k] * inv;
      ms_out[scoreBase + (size_t)t * LL + k] = p;
      sMS[t * 64 + k] = (bf16_t)p;
    }
  }
  __syncthreads();

  // ---- values: score @ V (bf16, K=64 -> two k-steps) ----
  v16bf a0 = frag_a_bf(sTS + m * 64, 64, lane);        // keys 0..31
  v16bf a1 = frag_a_bf(sTS + m * 64 + 32, 64, lane);   // keys 32..63
  v16bf p0 = frag_a_bf(sMS + m * 64, 64, lane);
  v16bf p1 = frag_a_bf(sMS + m * 64 + 32, 64, lane);
#pragma unroll
  for (int dt = 0; dt < 2; ++dt) {
    v16bf b0 = frag_b_bf(sTVt + dt * 16 * 64, 64, lane);       // keys 0..31
    v16bf b1 = frag_b_bf(sTVt + dt * 16 * 64 + 32, 64, lane);  // keys 32..63
    v8f accT = {0, 0, 0, 0, 0, 0, 0, 0};
    accT = wmma_bf16(a0, b0, accT);
    accT = wmma_bf16(a1, b1, accT);
    v16bf c0 = frag_b_bf(sGVt + dt * 16 * 64, 64, lane);
    v16bf c1 = frag_b_bf(sGVt + dt * 16 * 64 + 32, 64, lane);
    v8f accM = {0, 0, 0, 0, 0, 0, 0, 0};
    accM = wmma_bf16(p0, c0, accM);
    accM = wmma_bf16(p1, c1, accM);
    const int q0 = m + ((lane < 16) ? 0 : 8);
    const int d = dt * 16 + (lane & 15);
#pragma unroll
    for (int i = 0; i < 8; ++i) {
      size_t o = featBase + (size_t)(q0 + i) * DD + d;
      TVAL[o] = accT[i];
      MVAL[o] = accM[i];
    }
  }
}

// ======================================================================
// out = LayerNorm(P + R) * g + b ; 32 lanes per row, 8 rows per block
// ======================================================================
__global__ __launch_bounds__(256)
void ta_add_ln(const float* __restrict__ P, const float* __restrict__ R,
               const float* __restrict__ g, const float* __restrict__ b,
               float* __restrict__ out) {
  const int t = threadIdx.x;
  const int lane = t & 31;
  const int row = blockIdx.x * 8 + (t >> 5);
  const size_t base = (size_t)row * DD + lane * 8;
  float x[8];
  float s = 0.f;
#pragma unroll
  for (int j = 0; j < 8; ++j) { x[j] = P[base + j] + R[base + j]; s += x[j]; }
#pragma unroll
  for (int d = 16; d > 0; d >>= 1) s += __shfl_xor(s, d, 32);
  const float mean = s * (1.0f / DD);
  float v = 0.f;
#pragma unroll
  for (int j = 0; j < 8; ++j) { float dv = x[j] - mean; v += dv * dv; }
#pragma unroll
  for (int d = 16; d > 0; d >>= 1) v += __shfl_xor(v, d, 32);
  const float inv = rsqrtf(v * (1.0f / DD) + LN_EPS);
#pragma unroll
  for (int j = 0; j < 8; ++j) {
    int c = lane * 8 + j;
    out[base + j] = (x[j] - mean) * inv * g[c] + b[c];
  }
}

// ======================================================================
extern "C" void kernel_launch(void* const* d_in, const int* in_sizes, int n_in,
                              void* d_out, int out_size, void* d_ws, size_t ws_size,
                              hipStream_t stream) {
  const float* tfQ = (const float*)d_in[0];
  const float* gfQ = (const float*)d_in[1];
  const float* tfK = (const float*)d_in[2];
  const float* gfK = (const float*)d_in[3];
  const unsigned char* mask = (const unsigned char*)d_in[4];
  const float* t_Wq = (const float*)d_in[5];
  const float* t_bq = (const float*)d_in[6];
  const float* t_Wk = (const float*)d_in[7];
  const float* t_bk = (const float*)d_in[8];
  const float* t_Wv = (const float*)d_in[9];
  const float* t_bv = (const float*)d_in[10];
  const float* g_Wq = (const float*)d_in[11];
  const float* g_bq = (const float*)d_in[12];
  const float* g_Wk = (const float*)d_in[13];
  const float* g_bk = (const float*)d_in[14];
  const float* g_Wv = (const float*)d_in[15];
  const float* g_bv = (const float*)d_in[16];
  const float* time_out_W = (const float*)d_in[17];
  const float* time_out_b = (const float*)d_in[18];
  const float* target_out_W = (const float*)d_in[19];
  const float* target_out_b = (const float*)d_in[20];
  const float* time_ln_g = (const float*)d_in[21];
  const float* time_ln_b = (const float*)d_in[22];
  const float* target_ln_g = (const float*)d_in[23];
  const float* target_ln_b = (const float*)d_in[24];

  float* out = (float*)d_out;
  const size_t S  = (size_t)BB * NN * LL * DD;       // 8,388,608  per feature tensor
  const size_t SC = (size_t)BB * NN * HH * LL * LL;  // 16,777,216 per score tensor

  float* ws = (float*)d_ws;                          // 8*S floats + 8*64K halfs
  float* TQ   = ws + 0 * S;
  float* TK   = ws + 1 * S;
  float* TV   = ws + 2 * S;
  float* GQ   = ws + 3 * S;
  float* GK   = ws + 4 * S;
  float* GV   = ws + 5 * S;
  float* TVAL = ws + 6 * S;
  float* MVAL = ws + 7 * S;
  half_t* WT  = (half_t*)(ws + 8 * S);               // 8 transposed f16 weights
  const size_t WSZ = 256 * 256;

  float* time_score_o   = out;
  float* target_score_o = out + SC;
  float* merge_score_o  = out + 2 * SC;
  float* merge_attn_o   = out + 3 * SC;      // merge_attn_value
  float* time_attn_o    = out + 3 * SC + S;  // time_attn_value

  // transpose+convert the 8 weight matrices (tiny; enables TDM f16 staging)
  ta_prep_wt<<<256, 256, 0, stream>>>(t_Wq, WT + 0 * WSZ);
  ta_prep_wt<<<256, 256, 0, stream>>>(t_Wk, WT + 1 * WSZ);
  ta_prep_wt<<<256, 256, 0, stream>>>(t_Wv, WT + 2 * WSZ);
  ta_prep_wt<<<256, 256, 0, stream>>>(g_Wq, WT + 3 * WSZ);
  ta_prep_wt<<<256, 256, 0, stream>>>(g_Wk, WT + 4 * WSZ);
  ta_prep_wt<<<256, 256, 0, stream>>>(g_Wv, WT + 5 * WSZ);
  ta_prep_wt<<<256, 256, 0, stream>>>(time_out_W,   WT + 6 * WSZ);
  ta_prep_wt<<<256, 256, 0, stream>>>(target_out_W, WT + 7 * WSZ);

  dim3 gg(MROWS / 64);
  // input projections
  ta_gemm256<<<gg, 256, GEMM_LDS_BYTES, stream>>>(tfQ, WT + 0 * WSZ, t_bq, TQ);
  ta_gemm256<<<gg, 256, GEMM_LDS_BYTES, stream>>>(tfK, WT + 1 * WSZ, t_bk, TK);
  ta_gemm256<<<gg, 256, GEMM_LDS_BYTES, stream>>>(tfK, WT + 2 * WSZ, t_bv, TV);
  ta_gemm256<<<gg, 256, GEMM_LDS_BYTES, stream>>>(gfQ, WT + 3 * WSZ, g_bq, GQ);
  ta_gemm256<<<gg, 256, GEMM_LDS_BYTES, stream>>>(gfK, WT + 4 * WSZ, g_bk, GK);
  ta_gemm256<<<gg, 256, GEMM_LDS_BYTES, stream>>>(gfK, WT + 5 * WSZ, g_bv, GV);

  // fused dual attention per (b,n,h)
  ta_attn<<<BB * NN * HH, 128, 0, stream>>>(TQ, TK, TV, GQ, GK, GV, mask,
                                            time_score_o, target_score_o, merge_score_o,
                                            TVAL, MVAL);

  // output projections (reuse TQ/GQ buffers as projection scratch)
  ta_gemm256<<<gg, 256, GEMM_LDS_BYTES, stream>>>(TVAL, WT + 6 * WSZ, time_out_b, TQ);
  ta_gemm256<<<gg, 256, GEMM_LDS_BYTES, stream>>>(MVAL, WT + 7 * WSZ, target_out_b, GQ);

  // residual + layernorm -> final outputs
  ta_add_ln<<<MROWS / 8, 256, 0, stream>>>(TQ, tfQ, time_ln_g, time_ln_b, time_attn_o);
  ta_add_ln<<<MROWS / 8, 256, 0, stream>>>(GQ, gfQ, target_ln_g, target_ln_b, merge_attn_o);
}
